// SALADWrapper_62612033241254
// MI455X (gfx1250) — matrix operand order repaired and checked by reading, not compiled, not run on hardware
//
#include <hip/hip_runtime.h>
#include <hip/hip_bf16.h>
#include <math.h>

typedef unsigned short bf16s;                                   // bf16 storage
typedef __attribute__((ext_vector_type(16))) __bf16 v16bf;      // WMMA A/B frag
typedef __attribute__((ext_vector_type(8)))  float  v8f;        // WMMA C/D frag

#define B_   8
#define N_   8192
#define C_   512
#define K_   64
#define D_   128
#define TOK_ 256
#define M_TOTAL (B_*N_)          // 65536 rows
#define OUTW (TOK_ + D_*K_)      // 8448 output width per batch

#define LDS_PITCH 264            // 256-K slice + 8 pad -> dword bank stride 4 per row

__device__ __forceinline__ bf16s f2bf(float f) {
    unsigned u = __float_as_uint(f);
    u += 0x7fffu + ((u >> 16) & 1u);        // round-to-nearest-even
    return (bf16s)(u >> 16);
}

union FragBF { v16bf v; uint4 u[2]; };

// ------------------------------------------------------------------ utilities
__global__ void cvt_f32_bf16(const float* __restrict__ in, bf16s* __restrict__ out, int n) {
    int i = blockIdx.x * blockDim.x + threadIdx.x;
    if (i < n) out[i] = f2bf(in[i]);
}

__global__ void zero_f32(float* p, int n) {
    int i = blockIdx.x * blockDim.x + threadIdx.x;
    if (i < n) p[i] = 0.0f;
}

// ------------------------------------------------------------------ GeM pooling
__global__ void gem_partial(const float* __restrict__ x, const float* __restrict__ gem_w,
                            float* __restrict__ g_part) {
    int b = blockIdx.y;
    int nbase = blockIdx.x * 256;
    int t = threadIdx.x;
    float w = gem_w[0];
    float p = fminf(log1pf(expf(w)) + 1e-3f, 10.0f);
    float s0 = 0.f, s1 = 0.f;
    const float* xb = x + (size_t)b * N_ * C_;
    for (int n = nbase; n < nbase + 256; ++n) {
        const float* xr = xb + (size_t)n * C_;
        s0 += powf(fmaxf(xr[t],       1e-6f), p);
        s1 += powf(fmaxf(xr[t + 256], 1e-6f), p);
    }
    atomicAdd(&g_part[b * C_ + t],       s0);
    atomicAdd(&g_part[b * C_ + t + 256], s1);
}

__global__ void gem_final(const float* __restrict__ g_part, const float* __restrict__ gem_w,
                          float* __restrict__ g) {
    int i = blockIdx.x * blockDim.x + threadIdx.x;
    if (i >= B_ * C_) return;
    float w = gem_w[0];
    float p = fminf(log1pf(expf(w)) + 1e-3f, 10.0f);
    float v = powf(g_part[i] / (float)N_, 1.0f / p);
    if (!(v >= -3.402823466e38f && v <= 3.402823466e38f)) v = 0.0f;   // !isfinite -> 0
    g[i] = v;
}

// ------------------------------------------------------------------ token MLP (tiny, fp32)
__global__ void token_l1(const float* __restrict__ g, const float* __restrict__ W,
                         const float* __restrict__ bias, float* __restrict__ h) {
    int i = blockIdx.x * blockDim.x + threadIdx.x;   // b*512+o
    if (i >= B_ * 512) return;
    int b = i >> 9, o = i & 511;
    const float* gr = g + b * C_;
    const float* wr = W + (size_t)o * C_;
    float s = bias[o];
    for (int k = 0; k < C_; ++k) s += gr[k] * wr[k];
    h[i] = fmaxf(s, 0.0f);
}

__global__ void token_l2(const float* __restrict__ h, const float* __restrict__ W,
                         const float* __restrict__ bias, float* __restrict__ stage) {
    int i = blockIdx.x * blockDim.x + threadIdx.x;   // b*256+o
    if (i >= B_ * TOK_) return;
    int b = i >> 8, o = i & 255;
    const float* hr = h + b * 512;
    const float* wr = W + (size_t)o * 512;
    float s = bias[o];
    for (int k = 0; k < 512; ++k) s += hr[k] * wr[k];
    stage[(size_t)b * OUTW + o] = s;
}

// ------------------------------------------------------------------ WMMA GEMM:  out = act(A @ W^T + bias)
// A: [M, Kdim] bf16 row-major.  W: [Nout, Kdim] bf16 row-major.
// Block = 8 waves; wave tile 16(M) x 64(N); block tile 128 x 64.
// The block's 64-row W tile is staged into LDS in 256-wide K phases with
// global_load_async_to_lds_b128 (ASYNCcnt); B fragments then come from LDS.
// MODE 0: ReLU, store bf16 row-major [M, Nout]             (hidden layer H)
// MODE 1: store bf16 transposed to localT [B, D, N]        (local layer 2)
// MODE 2: store f32 transposed to S [B, K+1, N] rows 0..63 (score layer 2)
template<int MODE>
__global__ void __launch_bounds__(256)
gemm_wmma(const bf16s* __restrict__ A, const bf16s* __restrict__ W,
          const float* __restrict__ bias, void* __restrict__ out,
          int Kdim, int Nout) {
    __shared__ __align__(16) bf16s Wlds[64 * LDS_PITCH];   // 33.8 KB

    const int lane = threadIdx.x & 31;
    const int wave = threadIdx.x >> 5;
    const int m0   = blockIdx.x * 128 + wave * 16;
    const int n0   = blockIdx.y * 64;
    const int half = lane >> 4;       // 0/1
    const int lan  = lane & 15;

    const bf16s* Arow = A + (size_t)(m0 + lan) * Kdim;

    v8f acc[4] = {};
    const int aoff = half * 8;    // A frag: chunks at k0+aoff and k0+aoff+16
    const int boff = half * 8;    // B frag: chunks at k0+boff and k0+boff+16

    for (int kphase = 0; kphase < Kdim; kphase += 256) {
        // -------- cooperative async stage of W[n0:n0+64, kphase:kphase+256]
        #pragma unroll 1
        for (int c = threadIdx.x; c < 64 * 32; c += 256) {
            int r  = c >> 5;           // W row (output column) 0..63
            int cc = c & 31;           // 16-byte chunk within 256-K slice
            const bf16s* gp = W + (size_t)(n0 + r) * Kdim + kphase + cc * 8;
            unsigned lo = (unsigned)(uintptr_t)&Wlds[r * LDS_PITCH + cc * 8];
            asm volatile("global_load_async_to_lds_b128 %0, %1, off"
                         :: "v"(lo), "v"(gp) : "memory");
        }
        asm volatile("s_wait_asynccnt 0x0" ::: "memory");
        __syncthreads();

        // -------- WMMA over this K phase (A from global, B from LDS)
        for (int k0 = 0; k0 < 256; k0 += 32) {
            FragBF a;
            a.u[0] = *(const uint4*)(Arow + kphase + k0 + aoff);
            a.u[1] = *(const uint4*)(Arow + kphase + k0 + aoff + 16);
            #pragma unroll
            for (int t = 0; t < 4; ++t) {
                FragBF bfr;
                const bf16s* lp = &Wlds[(t * 16 + lan) * LDS_PITCH + k0 + boff];
                bfr.u[0] = *(const uint4*)(lp);
                bfr.u[1] = *(const uint4*)(lp + 16);
                acc[t] = __builtin_amdgcn_wmma_f32_16x16x32_bf16(
                    false, a.v, false, bfr.v, (short)0, acc[t], false, false);
            }
        }
        __syncthreads();   // protect LDS before next phase overwrites it
    }

    #pragma unroll
    for (int t = 0; t < 4; ++t) {
        int col = n0 + t * 16 + lan;
        float bv = bias[col];
        #pragma unroll
        for (int j = 0; j < 8; ++j) {
            int row = m0 + j + 8 * half;
            float val = acc[t][j] + bv;
            if (MODE == 0) {
                val = fmaxf(val, 0.0f);
                ((bf16s*)out)[(size_t)row * Nout + col] = f2bf(val);
            } else if (MODE == 1) {
                int b = row >> 13, n = row & (N_ - 1);
                ((bf16s*)out)[((size_t)b * D_ + col) * N_ + n] = f2bf(val);
            } else {
                int b = row >> 13, n = row & (N_ - 1);
                ((float*)out)[((size_t)b * (K_ + 1) + col) * N_ + n] = val;
            }
        }
    }
}

// ------------------------------------------------------------------ Sinkhorn
__global__ void dust_fill(float* __restrict__ S, const float* __restrict__ dust) {
    int i = blockIdx.x * blockDim.x + threadIdx.x;   // b*N + n
    if (i >= B_ * N_) return;
    int b = i >> 13, n = i & (N_ - 1);
    S[((size_t)b * (K_ + 1) + K_) * N_ + n] = dust[0];
}

__global__ void __launch_bounds__(256)
sk_u(const float* __restrict__ S, const float* __restrict__ v, float* __restrict__ u) {
    __shared__ float red[256];
    int bk = blockIdx.x;                      // b*(K+1)+k
    int b = bk / (K_ + 1), k = bk % (K_ + 1);
    int t = threadIdx.x;
    const float* row = S + (size_t)bk * N_;
    const float* vb  = v + (size_t)b * N_;
    float m = -__builtin_inff();
    for (int i = t; i < N_; i += 256) m = fmaxf(m, row[i] + vb[i]);
    red[t] = m; __syncthreads();
    for (int o = 128; o > 0; o >>= 1) { if (t < o) red[t] = fmaxf(red[t], red[t + o]); __syncthreads(); }
    float gm = red[0]; __syncthreads();
    float s = 0.f;
    for (int i = t; i < N_; i += 256) s += expf(row[i] + vb[i] - gm);
    red[t] = s; __syncthreads();
    for (int o = 128; o > 0; o >>= 1) { if (t < o) red[t] += red[t + o]; __syncthreads(); }
    if (t == 0) {
        float norm = -logf((float)(N_ + K_));
        float loga = norm + (k == K_ ? logf((float)(N_ - K_)) : 0.0f);
        u[bk] = loga - (gm + logf(red[0]));
    }
}

__global__ void sk_v(const float* __restrict__ S, const float* __restrict__ u,
                     float* __restrict__ v) {
    int i = blockIdx.x * blockDim.x + threadIdx.x;   // b*N + n
    if (i >= B_ * N_) return;
    int b = i >> 13, n = i & (N_ - 1);
    const float* Sb = S + (size_t)b * (K_ + 1) * N_ + n;
    const float* ub = u + b * (K_ + 1);
    float m = -__builtin_inff(), s = 0.f;
    for (int k = 0; k <= K_; ++k) {                  // online logsumexp over 65 rows
        float x = Sb[(size_t)k * N_] + ub[k];
        if (x > m) { s = s * expf(m - x) + 1.0f; m = x; }
        else       { s += expf(x - m); }
    }
    float norm = -logf((float)(N_ + K_));
    v[i] = norm - (m + logf(s));
}

__global__ void p_mat(const float* __restrict__ S, const float* __restrict__ u,
                      const float* __restrict__ v, bf16s* __restrict__ P) {
    int i = blockIdx.x * blockDim.x + threadIdx.x;   // ((b*K)+k)*N + n
    if (i >= B_ * K_ * N_) return;
    int n = i & (N_ - 1);
    int k = (i >> 13) & (K_ - 1);
    int b = i >> 19;
    float norm = -logf((float)(N_ + K_));
    float val = expf(S[((size_t)b * (K_ + 1) + k) * N_ + n] + u[b * (K_ + 1) + k]
                     + v[(size_t)b * N_ + n] - norm);
    P[i] = f2bf(val);
}

// ------------------------------------------------------------------ einsum via WMMA
// agg[b][d*64+k] = sum_n P[b][k][n] * localT[b][d][n]   (per-batch 64x128 GEMM, Kdim=N)
__global__ void __launch_bounds__(256)
einsum_wmma(const bf16s* __restrict__ P, const bf16s* __restrict__ Lt,
            float* __restrict__ stage) {
    int b = blockIdx.x;
    int lane = threadIdx.x & 31, wave = threadIdx.x >> 5;
    int half = lane >> 4, lan = lane & 15;
    int mtile = wave >> 1;            // k-tile 0..3  (M dim = K_ = 64)
    int nbase = (wave & 1) * 4;       // d-tiles nbase..nbase+3 (N dim = D_ = 128)
    const bf16s* Pb = P  + (size_t)b * K_ * N_;
    const bf16s* Lb = Lt + (size_t)b * D_ * N_;
    const bf16s* Arow = Pb + (size_t)(mtile * 16 + lan) * N_;
    const bf16s* Br[4];
    #pragma unroll
    for (int t = 0; t < 4; ++t) Br[t] = Lb + (size_t)((nbase + t) * 16 + lan) * N_;

    v8f acc[4] = {};
    const int aoff = half * 8;
    const int boff = half * 8;
    for (int k0 = 0; k0 < N_; k0 += 32) {
        FragBF a;
        a.u[0] = *(const uint4*)(Arow + k0 + aoff);
        a.u[1] = *(const uint4*)(Arow + k0 + aoff + 16);
        #pragma unroll
        for (int t = 0; t < 4; ++t) {
            FragBF bfr;
            bfr.u[0] = *(const uint4*)(Br[t] + k0 + boff);
            bfr.u[1] = *(const uint4*)(Br[t] + k0 + boff + 16);
            acc[t] = __builtin_amdgcn_wmma_f32_16x16x32_bf16(
                false, a.v, false, bfr.v, (short)0, acc[t], false, false);
        }
    }
    float* outb = stage + (size_t)b * OUTW + TOK_;
    #pragma unroll
    for (int t = 0; t < 4; ++t) {
        int d = (nbase + t) * 16 + lan;
        #pragma unroll
        for (int j = 0; j < 8; ++j) {
            int k = mtile * 16 + j + 8 * half;
            outb[d * K_ + k] = acc[t][j];
        }
    }
}

// ------------------------------------------------------------------ final 3-stage L2 norm
__global__ void __launch_bounds__(256)
final_norm(const float* __restrict__ stage, float* __restrict__ out) {
    __shared__ float red[256];
    __shared__ float shA, shB;
    int b = blockIdx.x, t = threadIdx.x;
    const float* sb = stage + (size_t)b * OUTW;
    float s1 = 0.f;
    for (int i = t; i < TOK_; i += 256) { float v = sb[i]; s1 += v * v; }
    red[t] = s1; __syncthreads();
    for (int o = 128; o > 0; o >>= 1) { if (t < o) red[t] += red[t + o]; __syncthreads(); }
    if (t == 0) shA = red[0];
    __syncthreads();
    float s2 = 0.f;
    for (int i = TOK_ + t; i < OUTW; i += 256) { float v = sb[i]; s2 += v * v; }
    red[t] = s2; __syncthreads();
    for (int o = 128; o > 0; o >>= 1) { if (t < o) red[t] += red[t + o]; __syncthreads(); }
    if (t == 0) shB = red[0];
    __syncthreads();
    float n1 = fmaxf(sqrtf(shA), 1e-12f);
    float n2 = fmaxf(sqrtf(shB), 1e-12f);
    float tot = shA / (n1 * n1) + shB / (n2 * n2);
    float n3 = fmaxf(sqrtf(tot), 1e-12f);
    float* ob = out + (size_t)b * OUTW;
    for (int i = t; i < OUTW; i += 256) {
        float pn = (i < TOK_) ? n1 : n2;
        ob[i] = sb[i] / (pn * n3);
    }
}

// ------------------------------------------------------------------ launcher
extern "C" void kernel_launch(void* const* d_in, const int* in_sizes, int n_in,
                              void* d_out, int out_size, void* d_ws, size_t ws_size,
                              hipStream_t stream) {
    const float* x     = (const float*)d_in[0];
    const float* gem_w = (const float*)d_in[1];
    const float* dust  = (const float*)d_in[2];
    const float* tW1 = (const float*)d_in[3];  const float* tb1 = (const float*)d_in[4];
    const float* tW2 = (const float*)d_in[5];  const float* tb2 = (const float*)d_in[6];
    const float* cW1 = (const float*)d_in[7];  const float* cb1 = (const float*)d_in[8];
    const float* cW2 = (const float*)d_in[9];  const float* cb2 = (const float*)d_in[10];
    const float* sW1 = (const float*)d_in[11]; const float* sb1 = (const float*)d_in[12];
    const float* sW2 = (const float*)d_in[13]; const float* sb2 = (const float*)d_in[14];
    float* outf = (float*)d_out;

    char* ws = (char*)d_ws;
    size_t off = 0;
    auto alloc = [&](size_t bytes) -> void* {
        void* p = ws + off;
        off += (bytes + 255) & ~(size_t)255;
        return p;
    };
    bf16s* x_bf  = (bf16s*)alloc((size_t)M_TOTAL * C_ * 2);       // 64 MiB
    bf16s* cW1b  = (bf16s*)alloc((size_t)512 * 512 * 2);
    bf16s* cW2b  = (bf16s*)alloc((size_t)128 * 512 * 2);
    bf16s* sW1b  = (bf16s*)alloc((size_t)512 * 512 * 2);
    bf16s* sW2b  = (bf16s*)alloc((size_t)64  * 512 * 2);
    bf16s* Hbf   = (bf16s*)alloc((size_t)M_TOTAL * 512 * 2);      // 64 MiB, reused by both branches
    bf16s* Ltb   = (bf16s*)alloc((size_t)B_ * D_ * N_ * 2);       // 16 MiB, localT
    float* Sbuf  = (float*)alloc((size_t)B_ * (K_ + 1) * N_ * 4); // 16.25 MiB
    float* ubuf  = (float*)alloc((size_t)B_ * (K_ + 1) * 4);
    float* vbuf  = (float*)alloc((size_t)B_ * N_ * 4);
    bf16s* Pbuf  = (bf16s*)alloc((size_t)B_ * K_ * N_ * 2);       // 8 MiB
    float* gpart = (float*)alloc((size_t)B_ * C_ * 4);
    float* gbuf  = (float*)alloc((size_t)B_ * C_ * 4);
    float* htok  = (float*)alloc((size_t)B_ * 512 * 4);
    float* stage = (float*)alloc((size_t)B_ * OUTW * 4);

    // precision conversions (once per call)
    int nx = M_TOTAL * C_;
    cvt_f32_bf16<<<(nx + 255) / 256, 256, 0, stream>>>(x, x_bf, nx);
    cvt_f32_bf16<<<(512 * 512 + 255) / 256, 256, 0, stream>>>(cW1, cW1b, 512 * 512);
    cvt_f32_bf16<<<(128 * 512 + 255) / 256, 256, 0, stream>>>(cW2, cW2b, 128 * 512);
    cvt_f32_bf16<<<(512 * 512 + 255) / 256, 256, 0, stream>>>(sW1, sW1b, 512 * 512);
    cvt_f32_bf16<<<(64  * 512 + 255) / 256, 256, 0, stream>>>(sW2, sW2b, 64 * 512);

    // GeM pooling + global token MLP (fp32, tiny)
    zero_f32<<<(B_ * C_ + 255) / 256, 256, 0, stream>>>(gpart, B_ * C_);
    gem_partial<<<dim3(N_ / 256, B_), 256, 0, stream>>>(x, gem_w, gpart);
    gem_final<<<(B_ * C_ + 255) / 256, 256, 0, stream>>>(gpart, gem_w, gbuf);
    token_l1<<<(B_ * 512 + 255) / 256, 256, 0, stream>>>(gbuf, tW1, tb1, htok);
    token_l2<<<(B_ * TOK_ + 255) / 256, 256, 0, stream>>>(htok, tW2, tb2, stage);

    // local branch: H = relu(x@cW1^T+cb1) ; localT = (H@cW2^T+cb2)^T
    gemm_wmma<0><<<dim3(M_TOTAL / 128, 512 / 64), 256, 0, stream>>>(x_bf, cW1b, cb1, Hbf, 512, 512);
    gemm_wmma<1><<<dim3(M_TOTAL / 128, 128 / 64), 256, 0, stream>>>(Hbf, cW2b, cb2, Ltb, 512, 128);
    // score branch: H = relu(x@sW1^T+sb1) ; S[0..63] = (H@sW2^T+sb2)^T
    gemm_wmma<0><<<dim3(M_TOTAL / 128, 512 / 64), 256, 0, stream>>>(x_bf, sW1b, sb1, Hbf, 512, 512);
    gemm_wmma<2><<<dim3(M_TOTAL / 128, 64 / 64), 256, 0, stream>>>(Hbf, sW2b, sb2, Sbuf, 512, 64);

    // Sinkhorn (3 iterations)
    dust_fill<<<(B_ * N_ + 255) / 256, 256, 0, stream>>>(Sbuf, dust);
    zero_f32<<<(B_ * N_ + 255) / 256, 256, 0, stream>>>(vbuf, B_ * N_);
    for (int it = 0; it < 3; ++it) {
        sk_u<<<B_ * (K_ + 1), 256, 0, stream>>>(Sbuf, vbuf, ubuf);
        sk_v<<<(B_ * N_ + 255) / 256, 256, 0, stream>>>(Sbuf, ubuf, vbuf);
    }
    p_mat<<<(B_ * K_ * N_ + 255) / 256, 256, 0, stream>>>(Sbuf, ubuf, vbuf, Pbuf);

    // aggregation einsum (WMMA) + final normalization
    einsum_wmma<<<B_, 256, 0, stream>>>(Pbuf, Ltb, stage);
    final_norm<<<B_, 256, 0, stream>>>(stage, outf);
}
